// NNBase_58583353917843
// MI455X (gfx1250) — compile-verified, matrix operand order checked by reading, not run on hardware
//
#include <hip/hip_runtime.h>
#include <hip/hip_bf16.h>

typedef __bf16 bf16;
typedef __attribute__((ext_vector_type(16))) __bf16 v16bf;
typedef __attribute__((ext_vector_type(8)))  __bf16 v8bf;
typedef __attribute__((ext_vector_type(8)))  float  v8f;

#define H_   1024
#define N_   128
#define T_   64
#define AA_  16
#define H2_  2048
#define H3_  3072
#define KP1_ 2080   // A+2H = 2064 padded up to multiple of 32

// ---------------------------------------------------------------------------
// Fragment load: 16-bit A/B operand, 16x32 (MxK) layout.
// lane = (g, m), g = lane>>4, m = lane&15.  Row/col index = m for both halves.
// elements 0..7  hold K = k0 + g*8 + 0..7
// elements 8..15 hold K = k0 + 16 + g*8 + 0..7
// Caller passes p = rowbase + k0 + g*8 (16-byte aligned).
// ---------------------------------------------------------------------------
__device__ __forceinline__ v16bf load_frag16(const bf16* p) {
  v8bf lo = *reinterpret_cast<const v8bf*>(p);
  v8bf hi = *reinterpret_cast<const v8bf*>(p + 16);
  v16bf f;
#pragma unroll
  for (int i = 0; i < 8; ++i) { f[i] = lo[i]; f[i + 8] = hi[i]; }
  return f;
}

__device__ __forceinline__ v8f wmma_bf16(v16bf a, v16bf b, v8f c) {
  return __builtin_amdgcn_wmma_f32_16x16x32_bf16(false, a, false, b,
                                                 (short)0, c, false, false);
}

__device__ __forceinline__ float sigmoidf_(float x) {
  return 1.0f / (1.0f + __expf(-x));
}

// ---------------------------------------------------------------------------
// fp32 -> bf16 conversions (weights / x staged once per call)
// ---------------------------------------------------------------------------
__global__ void k_to_bf16(const float* __restrict__ s, bf16* __restrict__ d, int n) {
  int i = blockIdx.x * blockDim.x + threadIdx.x;
  if (i < n) d[i] = (bf16)s[i];
}

__global__ void k_to_bf16_pad(const float* __restrict__ s, bf16* __restrict__ d,
                              int rows, int scols, int dcols) {
  int i = blockIdx.x * blockDim.x + threadIdx.x;
  if (i < rows * dcols) {
    int r = i / dcols, c = i - r * dcols;
    d[i] = (c < scols) ? (bf16)s[r * scols + c] : (bf16)0.0f;
  }
}

// ---------------------------------------------------------------------------
// Per-step GRU state prep: h0 = h_prev * mask_t  (fp32 copy + bf16 copy)
// ---------------------------------------------------------------------------
__global__ void k_h0_prep(const float* __restrict__ hprev,
                          const float* __restrict__ masks_t,
                          float* __restrict__ h0f, bf16* __restrict__ h0b) {
  int i = blockIdx.x * blockDim.x + threadIdx.x;
  if (i < N_ * H_) {
    float v = hprev[i] * masks_t[i >> 10];
    h0f[i] = v;
    h0b[i] = (bf16)v;
  }
}

// ---------------------------------------------------------------------------
// Fused GRU step, register-blocked 2 row tiles per wave.
// grid = (N/32 = 4, 16), block = (32, 4) -> 64 col tiles.
// Per K-chunk: 20 b128 loads, 12 WMMAs (B fragments reused across row tiles).
// ---------------------------------------------------------------------------
__global__ void __launch_bounds__(128)
k_gru_step(const bf16* __restrict__ xbf_t,   // [128 x 1024]
           const bf16* __restrict__ h0b,     // [128 x 1024]
           const float* __restrict__ h0f,    // [128 x 1024]
           const bf16* __restrict__ wih,     // [3072 x 1024] bf16
           const bf16* __restrict__ whh,     // [3072 x 1024] bf16
           const float* __restrict__ b_ih, const float* __restrict__ b_hh,
           float* __restrict__ hout)         // [128 x 1024]
{
  const int rb   = blockIdx.x * 32;                 // two 16-row tiles
  const int ct   = blockIdx.y * 4 + threadIdx.y;
  const int lane = threadIdx.x;
  const int g = lane >> 4, m = lane & 15;
  const int col = ct * 16 + m;

  const bf16* ax  = xbf_t + (size_t)(rb + m) * H_ + g * 8;
  const bf16* ah  = h0b   + (size_t)(rb + m) * H_ + g * 8;
  const bf16* wri = wih + (size_t)col * H_ + g * 8;
  const bf16* wzi = wih + (size_t)(H_  + col) * H_ + g * 8;
  const bf16* wni = wih + (size_t)(H2_ + col) * H_ + g * 8;
  const bf16* wrh = whh + (size_t)col * H_ + g * 8;
  const bf16* wzh = whh + (size_t)(H_  + col) * H_ + g * 8;
  const bf16* wnh = whh + (size_t)(H2_ + col) * H_ + g * 8;

  v8f ar[2] = {{}, {}}, az[2] = {{}, {}}, ain[2] = {{}, {}}, ahn[2] = {{}, {}};
  const size_t rstep = (size_t)16 * H_;

  for (int k0 = 0; k0 < H_; k0 += 32) {
    v16bf fx0 = load_frag16(ax + k0);
    v16bf fx1 = load_frag16(ax + rstep + k0);
    v16bf fh0 = load_frag16(ah + k0);
    v16bf fh1 = load_frag16(ah + rstep + k0);
    v16bf w;
    w = load_frag16(wri + k0); ar[0]  = wmma_bf16(fx0, w, ar[0]);  ar[1]  = wmma_bf16(fx1, w, ar[1]);
    w = load_frag16(wrh + k0); ar[0]  = wmma_bf16(fh0, w, ar[0]);  ar[1]  = wmma_bf16(fh1, w, ar[1]);
    w = load_frag16(wzi + k0); az[0]  = wmma_bf16(fx0, w, az[0]);  az[1]  = wmma_bf16(fx1, w, az[1]);
    w = load_frag16(wzh + k0); az[0]  = wmma_bf16(fh0, w, az[0]);  az[1]  = wmma_bf16(fh1, w, az[1]);
    w = load_frag16(wni + k0); ain[0] = wmma_bf16(fx0, w, ain[0]); ain[1] = wmma_bf16(fx1, w, ain[1]);
    w = load_frag16(wnh + k0); ahn[0] = wmma_bf16(fh0, w, ahn[0]); ahn[1] = wmma_bf16(fh1, w, ahn[1]);
  }

  const float br  = b_ih[col]       + b_hh[col];
  const float bz  = b_ih[H_  + col] + b_hh[H_  + col];
  const float bin = b_ih[H2_ + col];
  const float bhn = b_hh[H2_ + col];

#pragma unroll
  for (int rr = 0; rr < 2; ++rr) {
#pragma unroll
    for (int i = 0; i < 8; ++i) {
      int row = rb + rr * 16 + i + 8 * g;           // C/D layout: M = i + 8*g
      float r = sigmoidf_(ar[rr][i] + br);
      float z = sigmoidf_(az[rr][i] + bz);
      float n = tanhf(ain[rr][i] + bin + r * (ahn[rr][i] + bhn));
      float h0 = h0f[(size_t)row * H_ + col];
      hout[(size_t)row * H_ + col] = (1.0f - z) * n + z * h0;
    }
  }
}

// ---------------------------------------------------------------------------
// Per-step P-path input build: cur = [a*m (16) | mu_prev*m + x*(1-m) |
//                                     xprev*m + x*(1-m) | pad zeros] as bf16
// ---------------------------------------------------------------------------
__global__ void k_p_prep(const float* __restrict__ x_t,      // [128 x 1024]
                         const float* __restrict__ lo, int lo_ld,
                         const float* __restrict__ hi, int hi_ld,
                         const float* __restrict__ act_t,    // [128 x 16]
                         const float* __restrict__ masks_t,  // [128]
                         bf16* __restrict__ cur)             // [128 x 2080]
{
  int i = blockIdx.x * blockDim.x + threadIdx.x;
  if (i >= N_ * KP1_) return;
  int n = i / KP1_, c = i - n * KP1_;
  float mv = masks_t[n];
  float v;
  if (c < AA_) {
    v = act_t[n * AA_ + c] * mv;
  } else if (c < AA_ + H_) {
    int j = c - AA_;
    v = lo[(size_t)n * lo_ld + j] * mv + x_t[(size_t)n * H_ + j] * (1.0f - mv);
  } else if (c < AA_ + H2_) {
    int j = c - (AA_ + H_);
    v = hi[(size_t)n * hi_ld + j] * mv + x_t[(size_t)n * H_ + j] * (1.0f - mv);
  } else {
    v = 0.0f;
  }
  cur[i] = (bf16)v;
}

// ---------------------------------------------------------------------------
// WMMA GEMM, register-blocked 4 row tiles per wave.
// Y[M x Nc] = act(A_bf[M x K] @ W_bf[Nc x K]^T + bias)
// grid = (M/64, Nc/64), block = (32, 4).
// Per K-chunk: 10 b128 loads, 4 WMMAs (B fragment reused across row tiles).
// ---------------------------------------------------------------------------
__global__ void __launch_bounds__(128)
k_gemm_wmma(const bf16* __restrict__ A, int lda,
            const bf16* __restrict__ W, int ldw,
            const float* __restrict__ bias,
            float* __restrict__ outF, int ldo,
            bf16* __restrict__ outB, int ldob,
            int K, int relu)
{
  const int rb   = blockIdx.x * 64;                 // four 16-row tiles
  const int ct   = blockIdx.y * 4 + threadIdx.y;
  const int lane = threadIdx.x;
  const int g = lane >> 4, m = lane & 15;
  const int col = ct * 16 + m;

  const bf16* arow = A + (size_t)(rb + m) * lda + g * 8;
  const bf16* wrow = W + (size_t)col * ldw + g * 8;

  v8f acc[4] = {{}, {}, {}, {}};
  for (int k0 = 0; k0 < K; k0 += 32) {
    v16bf b = load_frag16(wrow + k0);
#pragma unroll
    for (int r = 0; r < 4; ++r) {
      v16bf a = load_frag16(arow + (size_t)(r * 16) * lda + k0);
      acc[r] = wmma_bf16(a, b, acc[r]);
    }
  }

  float bv = bias ? bias[col] : 0.0f;
#pragma unroll
  for (int r = 0; r < 4; ++r) {
#pragma unroll
    for (int i = 0; i < 8; ++i) {
      int row = rb + r * 16 + i + 8 * g;
      float v = acc[r][i] + bv;
      if (relu) v = fmaxf(v, 0.0f);
      if (outF) outF[(size_t)row * ldo + col] = v;
      if (outB) outB[(size_t)row * ldob + col] = (bf16)v;
    }
  }
}

// ---------------------------------------------------------------------------
// Dual-weight WMMA GEMM (mu & lv heads share the A operand).
// grid = (M/64, Nc/64), block = (32, 4).
// Per K-chunk: 12 b128 loads, 8 WMMAs.
// ---------------------------------------------------------------------------
__global__ void __launch_bounds__(128)
k_gemm_wmma_dual(const bf16* __restrict__ A, int lda,
                 const bf16* __restrict__ W0, const bf16* __restrict__ W1, int ldw,
                 const float* __restrict__ bias0, const float* __restrict__ bias1,
                 float* __restrict__ out0, float* __restrict__ out1, int ldo,
                 int K)
{
  const int rb   = blockIdx.x * 64;
  const int ct   = blockIdx.y * 4 + threadIdx.y;
  const int lane = threadIdx.x;
  const int g = lane >> 4, m = lane & 15;
  const int col = ct * 16 + m;

  const bf16* arow = A + (size_t)(rb + m) * lda + g * 8;
  const bf16* w0r  = W0 + (size_t)col * ldw + g * 8;
  const bf16* w1r  = W1 + (size_t)col * ldw + g * 8;

  v8f acc0[4] = {{}, {}, {}, {}};
  v8f acc1[4] = {{}, {}, {}, {}};
  for (int k0 = 0; k0 < K; k0 += 32) {
    v16bf b0 = load_frag16(w0r + k0);
    v16bf b1 = load_frag16(w1r + k0);
#pragma unroll
    for (int r = 0; r < 4; ++r) {
      v16bf a = load_frag16(arow + (size_t)(r * 16) * lda + k0);
      acc0[r] = wmma_bf16(a, b0, acc0[r]);
      acc1[r] = wmma_bf16(a, b1, acc1[r]);
    }
  }

  float bv0 = bias0[col], bv1 = bias1[col];
#pragma unroll
  for (int r = 0; r < 4; ++r) {
#pragma unroll
    for (int i = 0; i < 8; ++i) {
      int row = rb + r * 16 + i + 8 * g;
      out0[(size_t)row * ldo + col] = acc0[r][i] + bv0;
      out1[(size_t)row * ldo + col] = acc1[r][i] + bv1;
    }
  }
}

// ---------------------------------------------------------------------------
// Final outputs: gru_hxs = gru_acc[T-1];  p_hxs = [mu[T-1] | x[T-1]] per row
// ---------------------------------------------------------------------------
__global__ void k_finalize(const float* __restrict__ gru_last,
                           const float* __restrict__ mu_last,
                           const float* __restrict__ x_last,
                           float* __restrict__ gru_hxs,
                           float* __restrict__ p_hxs)
{
  int i = blockIdx.x * blockDim.x + threadIdx.x;
  if (i < N_ * H_) {
    gru_hxs[i] = gru_last[i];
    int n = i >> 10, j = i & (H_ - 1);
    p_hxs[(size_t)n * H2_ + j]      = mu_last[i];
    p_hxs[(size_t)n * H2_ + H_ + j] = x_last[i];
  }
}

// ---------------------------------------------------------------------------
extern "C" void kernel_launch(void* const* d_in, const int* in_sizes, int n_in,
                              void* d_out, int out_size, void* d_ws, size_t ws_size,
                              hipStream_t stream) {
  const float* x       = (const float*)d_in[0];
  const float* hxs_gru = (const float*)d_in[1];
  const float* hxs_p   = (const float*)d_in[2];
  const float* masks   = (const float*)d_in[3];
  const float* paoh    = (const float*)d_in[4];
  const float* w_ih    = (const float*)d_in[5];
  const float* w_hh    = (const float*)d_in[6];
  const float* b_ih    = (const float*)d_in[7];
  const float* b_hh    = (const float*)d_in[8];
  const float* p_w1    = (const float*)d_in[9];
  const float* p_b1    = (const float*)d_in[10];
  const float* p_w2    = (const float*)d_in[11];
  const float* p_b2    = (const float*)d_in[12];
  const float* mu_w    = (const float*)d_in[13];
  const float* mu_b    = (const float*)d_in[14];
  const float* lv_w    = (const float*)d_in[15];
  const float* lv_b    = (const float*)d_in[16];

  float* out      = (float*)d_out;
  float* gru_acc  = out;                                   // [T*N*H]
  float* gru_hxs  = gru_acc + (size_t)T_ * N_ * H_;        // [N*H]
  float* p_mu_out = gru_hxs + (size_t)N_ * H_;             // [T*N*H]
  float* p_lv_out = p_mu_out + (size_t)T_ * N_ * H_;       // [T*N*H]
  float* p_hxs    = p_lv_out + (size_t)T_ * N_ * H_;       // [N*2H]

  char* ws = (char*)d_ws;
  size_t off = 0;
  auto carve = [&](size_t bytes) -> void* {
    off = (off + 255) & ~(size_t)255;
    void* p = ws + off;
    off += bytes;
    return p;
  };
  bf16*  x_bf   = (bf16*)carve((size_t)T_ * N_ * H_ * 2);
  bf16*  wih_bf = (bf16*)carve((size_t)H3_ * H_ * 2);
  bf16*  whh_bf = (bf16*)carve((size_t)H3_ * H_ * 2);
  bf16*  pw1_bf = (bf16*)carve((size_t)H2_ * KP1_ * 2);
  bf16*  pw2_bf = (bf16*)carve((size_t)H_ * H2_ * 2);
  bf16*  muw_bf = (bf16*)carve((size_t)H_ * H_ * 2);
  bf16*  lvw_bf = (bf16*)carve((size_t)H_ * H_ * 2);
  float* h0f    = (float*)carve((size_t)N_ * H_ * 4);
  bf16*  h0bf   = (bf16*)carve((size_t)N_ * H_ * 2);
  bf16*  curbf  = (bf16*)carve((size_t)N_ * KP1_ * 2);
  bf16*  d1bf   = (bf16*)carve((size_t)N_ * H2_ * 2);
  bf16*  d2bf   = (bf16*)carve((size_t)N_ * H_ * 2);

  auto cdiv = [](int a, int b) { return (a + b - 1) / b; };

  // ---- one-time weight / x staging to bf16 (L2-resident working set) ----
  k_to_bf16<<<cdiv(T_ * N_ * H_, 256), 256, 0, stream>>>(x, x_bf, T_ * N_ * H_);
  k_to_bf16<<<cdiv(H3_ * H_, 256), 256, 0, stream>>>(w_ih, wih_bf, H3_ * H_);
  k_to_bf16<<<cdiv(H3_ * H_, 256), 256, 0, stream>>>(w_hh, whh_bf, H3_ * H_);
  k_to_bf16_pad<<<cdiv(H2_ * KP1_, 256), 256, 0, stream>>>(p_w1, pw1_bf, H2_, H2_ + AA_, KP1_);
  k_to_bf16<<<cdiv(H_ * H2_, 256), 256, 0, stream>>>(p_w2, pw2_bf, H_ * H2_);
  k_to_bf16<<<cdiv(H_ * H_, 256), 256, 0, stream>>>(mu_w, muw_bf, H_ * H_);
  k_to_bf16<<<cdiv(H_ * H_, 256), 256, 0, stream>>>(lv_w, lvw_bf, H_ * H_);

  const dim3 blk(32, 4);
  for (int t = 0; t < T_; ++t) {
    const float* masks_t = masks + (size_t)t * N_;
    const float* x_t     = x + (size_t)t * N_ * H_;
    const bf16*  xbf_t   = x_bf + (size_t)t * N_ * H_;

    // ---- GRU recurrence ----
    const float* hprev = (t == 0) ? hxs_gru : gru_acc + (size_t)(t - 1) * N_ * H_;
    k_h0_prep<<<cdiv(N_ * H_, 256), 256, 0, stream>>>(hprev, masks_t, h0f, h0bf);
    k_gru_step<<<dim3(N_ / 32, 16), blk, 0, stream>>>(
        xbf_t, h0bf, h0f, wih_bf, whh_bf, b_ih, b_hh,
        gru_acc + (size_t)t * N_ * H_);

    // ---- P recurrence ----
    const float* lo = (t == 0) ? hxs_p       : p_mu_out + (size_t)(t - 1) * N_ * H_;
    const float* hi = (t == 0) ? hxs_p + H_  : x + (size_t)(t - 1) * N_ * H_;
    int lo_ld = (t == 0) ? H2_ : H_;
    int hi_ld = (t == 0) ? H2_ : H_;
    k_p_prep<<<cdiv(N_ * KP1_, 256), 256, 0, stream>>>(
        x_t, lo, lo_ld, hi, hi_ld,
        paoh + (size_t)t * N_ * AA_, masks_t, curbf);

    // d1 = relu(cur @ p_w1.T + b1)   [128 x 2048]
    k_gemm_wmma<<<dim3(N_ / 64, H2_ / 64), blk, 0, stream>>>(
        curbf, KP1_, pw1_bf, KP1_, p_b1,
        (float*)nullptr, 0, d1bf, H2_, KP1_, 1);
    // d2 = relu(d1 @ p_w2.T + b2)    [128 x 1024]
    k_gemm_wmma<<<dim3(N_ / 64, H_ / 64), blk, 0, stream>>>(
        d1bf, H2_, pw2_bf, H2_, p_b2,
        (float*)nullptr, 0, d2bf, H_, H2_, 1);
    // mu = d2 @ mu_w.T + mu_b ; lv = d2 @ lv_w.T + lv_b   [128 x 1024] each
    k_gemm_wmma_dual<<<dim3(N_ / 64, H_ / 64), blk, 0, stream>>>(
        d2bf, H_, muw_bf, lvw_bf, H_, mu_b, lv_b,
        p_mu_out + (size_t)t * N_ * H_, p_lv_out + (size_t)t * N_ * H_, H_, H_);
  }

  k_finalize<<<cdiv(N_ * H_, 256), 256, 0, stream>>>(
      gru_acc + (size_t)(T_ - 1) * N_ * H_,
      p_mu_out + (size_t)(T_ - 1) * N_ * H_,
      x + (size_t)(T_ - 1) * N_ * H_,
      gru_hxs, p_hxs);
}
